// EquivariantBackbone_73229192397399
// MI455X (gfx1250) — matrix operand order repaired and checked by reading, not compiled, not run on hardware
//
#include <hip/hip_runtime.h>
#include <math.h>

typedef __attribute__((ext_vector_type(16))) __bf16 v16bf;
typedef __attribute__((ext_vector_type(8)))  float  v8f;

#define NN 512
#define NE 2048

// ---------------- workspace layout (float units) ----------------
constexpr long OFF_W3J = 0;                     // 27*125 -> pad 3392
constexpr long OFF_X0  = 3392;                  // 512*128
constexpr long OFF_XC1 = OFF_X0  + 65536;       // 512*1152
constexpr long OFF_XS1 = OFF_XC1 + 589824;      // 512*1152
constexpr long OFF_XC2 = OFF_XS1 + 589824;      // 512*576
constexpr long OFF_XS2 = OFF_XC2 + 294912;      // 512*576
constexpr long OFF_XC3 = OFF_XS2 + 294912;      // 512*288
constexpr long OFF_XS3 = OFF_XC3 + 147456;      // 512*288
constexpr long OFF_HV1 = OFF_XS3 + 147456;      // 2048*12
constexpr long OFF_HV2 = OFF_HV1 + 24576;
constexpr long OFF_HV3 = OFF_HV2 + 24576;
constexpr long OFF_SH  = OFF_HV3 + 24576;       // 2048*12
constexpr long OFF_Z   = OFF_SH  + 24576;       // max 4587520 (conv2)
constexpr long OFF_BF  = OFF_Z   + 4587520;     // bf16 region

// ---------------- W3J init (exact reference math, double) ----------------
__device__ double dfact(int n){ double r=1.0; for(int i=2;i<=n;++i) r*=(double)i; return r; }

__device__ double w3jc(int l1,int l2,int l3,int m1,int m2,int m3){
  if (m1+m2+m3 != 0) return 0.0;
  double pref = (((l1-l2-m3)&1) ? -1.0 : 1.0) *
    sqrt(dfact(l1+l2-l3)*dfact(l1-l2+l3)*dfact(-l1+l2+l3)/dfact(l1+l2+l3+1)
         *dfact(l1+m1)*dfact(l1-m1)*dfact(l2+m2)*dfact(l2-m2)*dfact(l3+m3)*dfact(l3-m3));
  double s = 0.0;
  for (int t=0;t<=l1+l2-l3;++t){
    int d0=t, d1=l3-l2+t+m1, d2=l3-l1+t-m2, d3=l1+l2-l3-t, d4=l1-t-m1, d5=l2-t+m2;
    if (d0<0||d1<0||d2<0||d3<0||d4<0||d5<0) continue;
    double den = dfact(d0)*dfact(d1)*dfact(d2)*dfact(d3)*dfact(d4)*dfact(d5);
    s += ((t&1)?-1.0:1.0)/den;
  }
  return pref*s;
}

__device__ void urealm(int l, double Ur[5][5], double Ui[5][5]){
  for (int a=0;a<5;++a) for (int b=0;b<5;++b){ Ur[a][b]=0.0; Ui[a][b]=0.0; }
  Ur[l][l] = 1.0;
  const double is2 = 0.7071067811865476;
  for (int m=1;m<=l;++m){
    double p = (m&1)? -1.0:1.0;
    Ur[l+m][l+m] = p*is2;
    Ur[l+m][l-m] = is2;
    Ui[l-m][l-m] = is2;
    Ui[l-m][l+m] = -p*is2;
  }
}

__global__ void k_w3j(float* __restrict__ w3){
  int t = threadIdx.x;
  if (t >= 27) return;
  int l1 = t/9, l2 = (t/3)%3, l3 = t%3;
  float* dstp = w3 + t*125;
  for (int i=0;i<125;++i) dstp[i] = 0.f;
  int lo = (l1>l2)? (l1-l2):(l2-l1);
  if (l3 < lo || l3 > l1+l2) return;
  int d1=2*l1+1, d2=2*l2+1, d3=2*l3+1;
  double W[5][5][5];
  for (int a=0;a<d1;++a) for (int b=0;b<d2;++b) for (int c=0;c<d3;++c)
    W[a][b][c] = w3jc(l1,l2,l3, a-l1, b-l2, c-l3);
  double U1r[5][5],U1i[5][5],U2r[5][5],U2i[5][5],U3r[5][5],U3i[5][5];
  urealm(l1,U1r,U1i); urealm(l2,U2r,U2i); urealm(l3,U3r,U3i);
  double Cv[5][5][5]; double nrm=0.0;
  for (int a=0;a<d1;++a) for (int b=0;b<d2;++b) for (int c=0;c<d3;++c){
    double re=0.0, im=0.0;
    for (int mm=0;mm<d1;++mm) for (int nn=0;nn<d2;++nn) for (int oo=0;oo<d3;++oo){
      double w = W[mm][nn][oo];
      if (w == 0.0) continue;
      double pr = U1r[a][mm]*U2r[b][nn] - U1i[a][mm]*U2i[b][nn];
      double pi = U1r[a][mm]*U2i[b][nn] + U1i[a][mm]*U2r[b][nn];
      double qr = pr*U3r[c][oo] - pi*U3i[c][oo];
      double qi = pr*U3i[c][oo] + pi*U3r[c][oo];
      re += qr*w; im += qi*w;
    }
    double v = re + im;
    Cv[a][b][c] = v; nrm += v*v;
  }
  nrm = sqrt(nrm);
  double sc = (nrm > 0.0) ? 1.0/nrm : 0.0;
  for (int a=0;a<d1;++a) for (int b=0;b<d2;++b) for (int c=0;c<d3;++c)
    dstp[(a*5+b)*5+c] = (float)(Cv[a][b][c]*sc);
}

// ---------------- edge geometry + radial MLP hidden ----------------
__global__ void k_edge_prep(const float* __restrict__ pos, const int* __restrict__ src,
                            const int* __restrict__ dst,
                            const float* __restrict__ w1a, const float* __restrict__ w1b,
                            const float* __restrict__ w1c,
                            float* __restrict__ sh, float* __restrict__ hv1,
                            float* __restrict__ hv2, float* __restrict__ hv3){
  int e = blockIdx.x*64 + threadIdx.x;
  if (e >= NE) return;
  int s = src[e], d = dst[e];
  float vx = pos[s*3+0]-pos[d*3+0];
  float vy = pos[s*3+1]-pos[d*3+1];
  float vz = pos[s*3+2]-pos[d*3+2];
  float dist = sqrtf(vx*vx+vy*vy+vz*vz);
  float inv = 1.f / fmaxf(dist, 1e-12f);
  float x = vx*inv, y = vy*inv, zz = vz*inv;
  const float S3=1.7320508f, S15=3.8729833f, S5=2.2360680f;
  float* s9 = sh + (size_t)e*12;
  s9[0]=1.f; s9[1]=S3*y; s9[2]=S3*zz; s9[3]=S3*x;
  s9[4]=S15*x*y; s9[5]=S15*y*zz; s9[6]=0.5f*S5*(3.f*zz*zz-1.f);
  s9[7]=S15*x*zz; s9[8]=0.5f*S15*(x*x-y*y);
  s9[9]=0.f; s9[10]=0.f; s9[11]=0.f;
  float rb[11];
#pragma unroll
  for (int i=0;i<11;++i){ float t = (dist - 0.8f*i)*1.25f; rb[i] = expf(-t*t)*0.89285714f; }
  const float* wsl[3] = {w1a,w1b,w1c};
  float* hvl[3] = {hv1,hv2,hv3};
  for (int L=0;L<3;++L){
    const float* w1 = wsl[L]; float* hv = hvl[L];
    for (int j=0;j<12;++j){
      float acc=0.f;
#pragma unroll
      for (int i=0;i<11;++i) acc += rb[i]*w1[i*12+j];
      acc *= 0.30151134f;                  // 1/sqrt(11)
      hv[(size_t)e*12+j] = fmaxf(acc,0.f)*1.41421356f; // relu * sqrt(2)
    }
  }
}

// ---------------- B repack (fp32 -> bf16, K-contiguous transposed) ----------------
__global__ void k_repack_rw2(const float* __restrict__ w2, unsigned short* __restrict__ BTu,
                             int K, int mul, int mulShift, int C, int W2, int pbase,
                             int nCols, float scale){
  long tid = (long)blockIdx.x*256 + threadIdx.x;
  if (tid >= (long)nCols*K) return;
  int n = (int)(tid / K), k = (int)(tid - (long)n*K);
  int h = k >> mulShift, u = k & (mul-1);
  int p = n / C, w = n - p*C;
  float v = w2[(size_t)h*W2 + (size_t)(pbase+p)*mul*C + (size_t)u*C + w] * scale;
  reinterpret_cast<__bf16*>(BTu)[(size_t)n*K + k] = (__bf16)v;
}

__global__ void k_repack_si(const float* __restrict__ Wm, unsigned short* __restrict__ BTu,
                            int K, int N, float scale){
  int tid = blockIdx.x*256 + threadIdx.x;
  if (tid >= N*K) return;
  int n = tid / K, k = tid - n*K;
  reinterpret_cast<__bf16*>(BTu)[(size_t)n*K + k] = (__bf16)(Wm[(size_t)k*N + n] * scale);
}

// ---------------- fused WMMA GEMM ----------------
// A[(e,i),(h,u)] = hvec[e,h] * xin[node(e), xoff + u*d1 + i]   (USEH)
// A[(n,i),  u ]  =             xin[n,       xoff + u*d1 + i]   (!USEH)
// B = K-contiguous transposed bf16 weights, out = A@B with f32 accumulation.
#define XPAD 132   // LDS row stride (floats): 16B aligned rows, conflict-free b128 reads

template<bool USEH>
__global__ __launch_bounds__(64)
void k_gemm(const float* __restrict__ xin, int xstride, int xoff,
            const int* __restrict__ nodeIdx,
            const float* __restrict__ hvec,
            const unsigned short* __restrict__ BTu, int K,
            int mul, int mulShift, int d1,
            float* __restrict__ out,
            int zld, int siMode, int ostride, int ooff){
  __shared__ __align__(16) float x_sh[16*XPAD];
  __shared__ float hv_sh[16*13];
  __shared__ int rowNode[16], rowI[16], rowE[16];
  const int row0 = blockIdx.x * 16;
  if (threadIdx.x < 16) {
    int row = row0 + (int)threadIdx.x;
    int e = row / d1;
    int i = row - e*d1;
    rowE[threadIdx.x] = e;
    rowI[threadIdx.x] = i;
    rowNode[threadIdx.x] = nodeIdx ? nodeIdx[e] : e;
  }
  __syncthreads();
  for (int idx = threadIdx.x; idx < (16<<mulShift); idx += 64) {
    int r = idx >> mulShift, u = idx & (mul-1);
    x_sh[r*XPAD+u] = xin[(size_t)rowNode[r]*xstride + xoff + u*d1 + rowI[r]];
  }
  if (USEH) {
    for (int idx = threadIdx.x; idx < 16*12; idx += 64) {
      int r = idx / 12, j = idx - r*12;
      hv_sh[r*13+j] = hvec[(size_t)rowE[r]*12 + j];
    }
  }
  __syncthreads();

  const int lane = threadIdx.x & 31;
  const int wv   = threadIdx.x >> 5;
  const int m    = lane & 15;
  const int hi   = lane >> 4;
  const int ncol = blockIdx.y*32 + wv*16 + m;
  const __bf16* BT = reinterpret_cast<const __bf16*>(BTu);
  const __bf16* bp = BT + (size_t)ncol*K + hi*16;
  const float* xr = x_sh + m*XPAD;
  const float* hr = hv_sh + m*13;
  v8f acc = {0.f,0.f,0.f,0.f,0.f,0.f,0.f,0.f};
  for (int kc = 0; kc < K; kc += 32) {
    v16bf b = *(const v16bf*)(bp + kc);     // one contiguous 32B frag per lane
    // h index is uniform across a 32-wide, 32-aligned K chunk (mul % 32 == 0)
    // A-side x values are one contiguous 16-float run -> 4x ds_load_b128
    int ub = (USEH ? (kc & (mul-1)) : kc) + (hi ? 16 : 0);
    float4 p0 = *(const float4*)(xr + ub);
    float4 p1 = *(const float4*)(xr + ub + 4);
    float4 p2 = *(const float4*)(xr + ub + 8);
    float4 p3 = *(const float4*)(xr + ub + 12);
    if (USEH) {
      float hval = hr[kc >> mulShift];
      p0.x*=hval; p0.y*=hval; p0.z*=hval; p0.w*=hval;
      p1.x*=hval; p1.y*=hval; p1.z*=hval; p1.w*=hval;
      p2.x*=hval; p2.y*=hval; p2.z*=hval; p2.w*=hval;
      p3.x*=hval; p3.y*=hval; p3.z*=hval; p3.w*=hval;
    }
    v16bf a;
    a[0] =(__bf16)p0.x; a[1] =(__bf16)p0.y; a[2] =(__bf16)p0.z; a[3] =(__bf16)p0.w;
    a[4] =(__bf16)p1.x; a[5] =(__bf16)p1.y; a[6] =(__bf16)p1.z; a[7] =(__bf16)p1.w;
    a[8] =(__bf16)p2.x; a[9] =(__bf16)p2.y; a[10]=(__bf16)p2.z; a[11]=(__bf16)p2.w;
    a[12]=(__bf16)p3.x; a[13]=(__bf16)p3.y; a[14]=(__bf16)p3.z; a[15]=(__bf16)p3.w;
    acc = __builtin_amdgcn_wmma_f32_16x16x32_bf16(false, a, false, b, (short)0, acc, false, false);
  }
#pragma unroll
  for (int v = 0; v < 8; ++v) {
    int rr = v + (hi ? 8 : 0);
    float val = acc[v];
    if (siMode) {
      out[(size_t)rowNode[rr]*ostride + ooff + ncol*d1 + rowI[rr]] = val;
    } else {
      out[(size_t)(row0+rr)*zld + ncol] = val;
    }
  }
}

// ---------------- conv post: z * sh * W3J -> scatter-add to nodes ----------------
struct PathT { short blk, l1, l2, l3, pl; };
__constant__ PathT PTH1[3] = {{0,0,0,0,0},{0,0,1,1,1},{0,0,2,2,2}};
__constant__ PathT PTH2[11] = {
  {0,0,0,0,0},{0,0,1,1,1},{0,0,2,2,2},
  {1,1,0,1,0},{1,1,1,0,1},{1,1,1,2,2},{1,1,2,1,3},
  {2,2,0,2,0},{2,2,1,1,1},{2,2,2,0,2},{2,2,2,2,3}};

struct PostArgs {
  int C, ostride, nP, use11;
  long zoff0, zoff1, zoff2;
  int zld0, zld1, zld2;
  float a0, a1, a2;
};

__global__ void k_post(const float* __restrict__ z, const float* __restrict__ sh,
                       const int* __restrict__ srcIdx, const float* __restrict__ w3,
                       float* __restrict__ nodeOut, PostArgs A){
  int tid = blockIdx.x*blockDim.x + threadIdx.x;
  if (tid >= NE * A.C) return;
  int e = tid / A.C, w = tid - e*A.C;
  float shl[9];
#pragma unroll
  for (int j=0;j<9;++j) shl[j] = sh[(size_t)e*12+j];
  float o0 = 0.f, o1[3] = {0,0,0}, o2[5] = {0,0,0,0,0};
  long zoffs[3] = {A.zoff0, A.zoff1, A.zoff2};
  int zlds[3] = {A.zld0, A.zld1, A.zld2};
  float alph[3] = {A.a0, A.a1, A.a2};
  const PathT* P = A.use11 ? PTH2 : PTH1;
  for (int p = 0; p < A.nP; ++p) {
    PathT pt = P[p];
    int l1=pt.l1, l2=pt.l2, l3=pt.l3;
    int d1=2*l1+1, dj=2*l2+1, dk=2*l3+1;
    const float* zb = z + zoffs[pt.blk];
    int ld = zlds[pt.blk];
    int colb = pt.pl * A.C + w;
    float zi[5];
    for (int i=0;i<d1;++i) zi[i] = zb[(size_t)(e*d1+i)*ld + colb];
    const float* W = w3 + (((l1*3+l2)*3)+l3)*125;
    int joff = l2*l2;                // SH_OFF = l^2
    float al = alph[l3];
    for (int i=0;i<d1;++i)
      for (int j=0;j<dj;++j) {
        float zs = zi[i]*shl[joff+j]*al;
        for (int k=0;k<dk;++k) {
          float v = zs * W[(i*5+j)*5+k];
          if (l3==0) o0 += v; else if (l3==1) o1[k]+=v; else o2[k]+=v;
        }
      }
  }
  int s = srcIdx[e];
  float* nb = nodeOut + (size_t)s*A.ostride;
  int C = A.C;
  atomicAdd(&nb[w], o0);
#pragma unroll
  for (int k=0;k<3;++k) atomicAdd(&nb[C + w*3 + k], o1[k]);
#pragma unroll
  for (int k=0;k<5;++k) atomicAdd(&nb[4*C + w*5 + k], o2[k]);
}

// ---------------- pointwise ops ----------------
__global__ void k_zero(float* __restrict__ p, long n){
  long i = (long)blockIdx.x*256 + threadIdx.x;
  if (i < n) p[i] = 0.f;
}

__global__ void k_norm(float* __restrict__ x, int width){
  int n = blockIdx.x*64 + threadIdx.x;
  if (n >= NN) return;
  float* row = x + (size_t)n*width;
  float ss = 0.f;
  for (int i=0;i<width;++i){ float v=row[i]; ss += v*v; }
  float sc = 1.f/(sqrtf(ss)+1e-6f);
  for (int i=0;i<width;++i) row[i]*=sc;
}

__device__ __forceinline__ float softplusf(float x){ return (x > 20.f) ? x : log1pf(expf(x)); }

__global__ void k_nonlin(float* __restrict__ x, const float* __restrict__ b, int mul){
  int tid = blockIdx.x*256+threadIdx.x;
  if (tid >= NN*mul) return;
  int n = tid / mul, u = tid - n*mul;
  float b0=b[0], b1=b[1], b2=b[2];
  float* row = x + (size_t)n*(9*mul);
  row[u] = softplusf(row[u]+b0);
  float* v1 = row + mul + u*3;
  float ss = v1[0]*v1[0]+v1[1]*v1[1]+v1[2]*v1[2];
  float g = softplusf(sqrtf(ss + 1e-24f)+b1);
  v1[0]*=g; v1[1]*=g; v1[2]*=g;
  float* v2 = row + 4*mul + u*5;
  ss = v2[0]*v2[0]+v2[1]*v2[1]+v2[2]*v2[2]+v2[3]*v2[3]+v2[4]*v2[4];
  g = softplusf(sqrtf(ss + 1e-24f)+b2);
#pragma unroll
  for (int k=0;k<5;++k) v2[k]*=g;
}

// ---------------- host ----------------
extern "C" void kernel_launch(void* const* d_in, const int* in_sizes, int n_in,
                              void* d_out, int out_size, void* d_ws, size_t ws_size,
                              hipStream_t stream) {
  (void)in_sizes; (void)n_in; (void)out_size; (void)ws_size;
  const float* pos  = (const float*)d_in[0];
  const float* feat = (const float*)d_in[1];
  const int*   eidx = (const int*)d_in[2];
  const int*   srcI = eidx;
  const int*   dstI = eidx + NE;
  const float* si0w = (const float*)d_in[3];
  const float* c1w1 = (const float*)d_in[4];
  const float* c1w2 = (const float*)d_in[5];
  const float* si1w = (const float*)d_in[6];
  const float* nl1b = (const float*)d_in[7];
  const float* c2w1 = (const float*)d_in[8];
  const float* c2w2 = (const float*)d_in[9];
  const float* si2w = (const float*)d_in[10];
  const float* nl2b = (const float*)d_in[11];
  const float* c3w1 = (const float*)d_in[12];
  const float* c3w2 = (const float*)d_in[13];
  const float* si3w = (const float*)d_in[14];
  const float* nl3b = (const float*)d_in[15];
  const float* fsiw = (const float*)d_in[16];

  float* ws  = (float*)d_ws;
  float* W3  = ws + OFF_W3J;
  float* X0  = ws + OFF_X0;
  float* XC1 = ws + OFF_XC1;  float* XS1 = ws + OFF_XS1;
  float* XC2 = ws + OFF_XC2;  float* XS2 = ws + OFF_XS2;
  float* XC3 = ws + OFF_XC3;  float* XS3 = ws + OFF_XS3;
  float* HV1 = ws + OFF_HV1;  float* HV2 = ws + OFF_HV2;  float* HV3 = ws + OFF_HV3;
  float* SH  = ws + OFF_SH;
  float* Zb  = ws + OFF_Z;
  unsigned short* BF  = (unsigned short*)(ws + OFF_BF);
  unsigned short* B1  = BF;
  unsigned short* B2  = B1  + 589824;    // 384*1536
  unsigned short* B3  = B2  + 1081344;   // 704*1536
  unsigned short* Bs0 = B3  + 270336;    // 352*768
  unsigned short* Bs1 = Bs0 + 8192;
  unsigned short* Bs2 = Bs1 + 49152;
  unsigned short* Bs3 = Bs2 + 12288;
  unsigned short* Bf  = Bs3 + 3072;

  const float s12 = 0.28867513f; // 1/sqrt(12)

  // init: W3J tables + edge geometry/radial hidden + weight repacks
  k_w3j<<<1,32,0,stream>>>(W3);
  k_edge_prep<<<(NE+63)/64,64,0,stream>>>(pos, srcI, dstI, c1w1, c2w1, c3w1, SH, HV1, HV2, HV3);

  k_repack_rw2<<<(384*1536+255)/256,256,0,stream>>>(c1w2, B1, 1536,128,7,128, 49152, 0, 384, s12);
  k_repack_rw2<<<(192*1536+255)/256,256,0,stream>>>(c2w2, B2,            1536,128,7,64, 90112, 0, 192, s12);
  k_repack_rw2<<<(256*1536+255)/256,256,0,stream>>>(c2w2, B2+192*1536,   1536,128,7,64, 90112, 3, 256, s12);
  k_repack_rw2<<<(256*1536+255)/256,256,0,stream>>>(c2w2, B2+448*1536,   1536,128,7,64, 90112, 7, 256, s12);
  k_repack_rw2<<<( 96*768+255)/256,256,0,stream>>>(c3w2, B3,             768,64,6,32, 22528, 0,  96, s12);
  k_repack_rw2<<<(128*768+255)/256,256,0,stream>>>(c3w2, B3+96*768,      768,64,6,32, 22528, 3, 128, s12);
  k_repack_rw2<<<(128*768+255)/256,256,0,stream>>>(c3w2, B3+224*768,     768,64,6,32, 22528, 7, 128, s12);

  k_repack_si<<<(128*64+255)/256,256,0,stream>>>(si0w, Bs0, 64, 128, 0.125f);               // 1/sqrt(64)
  for (int l=0;l<3;++l)
    k_repack_si<<<(128*128+255)/256,256,0,stream>>>(si1w+l*16384, Bs1+l*16384, 128,128, 0.08838835f);
  for (int l=0;l<3;++l)
    k_repack_si<<<(64*64+255)/256,256,0,stream>>>(si2w+l*4096, Bs2+l*4096, 64,64, 0.125f);
  for (int l=0;l<3;++l)
    k_repack_si<<<(32*32+255)/256,256,0,stream>>>(si3w+l*1024, Bs3+l*1024, 32,32, 0.17677670f);
  k_repack_si<<<(32*32+255)/256,256,0,stream>>>(fsiw, Bf, 32,32, 0.17677670f);

  // x0 = features @ si0_w / sqrt(64)
  k_gemm<false><<<dim3(NN/16,4),64,0,stream>>>(feat, 64, 0, nullptr, nullptr, Bs0, 64, 64, 6, 1,
                                               X0, 0, 1, 128, 0);

  // ---- conv1 (in_ls=[0], mul=128, C=128) ----
  k_zero<<<(512L*1152+255)/256,256,0,stream>>>(XC1, 512L*1152);
  k_gemm<true><<<dim3(NE/16,12),64,0,stream>>>(X0, 128, 0, dstI, HV1, B1, 1536, 128, 7, 1,
                                               Zb, 384, 0, 0, 0);
  {
    PostArgs A = {128, 1152, 3, 0, 0,0,0, 384,0,0,
                  1.f/sqrtf(128.f), sqrtf(3.f/128.f), sqrtf(5.f/128.f)};
    k_post<<<(NE*128)/256,256,0,stream>>>(Zb, SH, srcI, W3, XC1, A);
  }
  k_norm<<<NN/64,64,0,stream>>>(XC1, 1152);
  for (int l=0;l<3;++l) {
    int d = 2*l+1;
    k_gemm<false><<<dim3(NN*d/16,4),64,0,stream>>>(XC1, 1152, 128*l*l, nullptr, nullptr,
                                                   Bs1+l*16384, 128, 128, 7, d,
                                                   XS1, 0, 1, 1152, 128*l*l);
  }
  k_nonlin<<<(NN*128)/256,256,0,stream>>>(XS1, nl1b, 128);

  // ---- conv2 (in_ls=[0,1,2], mul=128, C=64) ----
  k_zero<<<(512L*576+255)/256,256,0,stream>>>(XC2, 512L*576);
  k_gemm<true><<<dim3(NE/16,6),64,0,stream>>>(XS1, 1152, 0,   dstI, HV2, B2,           1536, 128, 7, 1, Zb,          192, 0,0,0);
  k_gemm<true><<<dim3(NE*3/16,8),64,0,stream>>>(XS1, 1152, 128, dstI, HV2, B2+192*1536, 1536, 128, 7, 3, Zb+393216,  256, 0,0,0);
  k_gemm<true><<<dim3(NE*5/16,8),64,0,stream>>>(XS1, 1152, 512, dstI, HV2, B2+448*1536, 1536, 128, 7, 5, Zb+1966080, 256, 0,0,0);
  {
    PostArgs A = {64, 576, 11, 1, 0,393216,1966080, 192,256,256,
                  1.f/sqrtf(384.f), sqrtf(3.f/512.f), sqrtf(5.f/512.f)};
    k_post<<<(NE*64)/256,256,0,stream>>>(Zb, SH, srcI, W3, XC2, A);
  }
  k_norm<<<NN/64,64,0,stream>>>(XC2, 576);
  for (int l=0;l<3;++l) {
    int d = 2*l+1;
    k_gemm<false><<<dim3(NN*d/16,2),64,0,stream>>>(XC2, 576, 64*l*l, nullptr, nullptr,
                                                   Bs2+l*4096, 64, 64, 6, d,
                                                   XS2, 0, 1, 576, 64*l*l);
  }
  k_nonlin<<<(NN*64)/256,256,0,stream>>>(XS2, nl2b, 64);

  // ---- conv3 (in_ls=[0,1,2], mul=64, C=32) ----
  k_zero<<<(512L*288+255)/256,256,0,stream>>>(XC3, 512L*288);
  k_gemm<true><<<dim3(NE/16,3),64,0,stream>>>(XS2, 576, 0,   dstI, HV3, B3,          768, 64, 6, 1, Zb,         96, 0,0,0);
  k_gemm<true><<<dim3(NE*3/16,4),64,0,stream>>>(XS2, 576, 64,  dstI, HV3, B3+96*768,  768, 64, 6, 3, Zb+196608, 128, 0,0,0);
  k_gemm<true><<<dim3(NE*5/16,4),64,0,stream>>>(XS2, 576, 256, dstI, HV3, B3+224*768, 768, 64, 6, 5, Zb+983040, 128, 0,0,0);
  {
    PostArgs A = {32, 288, 11, 1, 0,196608,983040, 96,128,128,
                  1.f/sqrtf(192.f), sqrtf(3.f/256.f), sqrtf(5.f/256.f)};
    k_post<<<(NE*32)/256,256,0,stream>>>(Zb, SH, srcI, W3, XC3, A);
  }
  k_norm<<<NN/64,64,0,stream>>>(XC3, 288);
  for (int l=0;l<3;++l) {
    int d = 2*l+1;
    k_gemm<false><<<dim3(NN*d/16,1),64,0,stream>>>(XC3, 288, 32*l*l, nullptr, nullptr,
                                                   Bs3+l*1024, 32, 32, 5, d,
                                                   XS3, 0, 1, 288, 32*l*l);
  }
  k_nonlin<<<(NN*32)/256,256,0,stream>>>(XS3, nl3b, 32);

  // final self-interaction: only the l=0 block feeds the 512x32 output
  k_gemm<false><<<dim3(NN/16,1),64,0,stream>>>(XS3, 288, 0, nullptr, nullptr, Bf, 32, 32, 5, 1,
                                               (float*)d_out, 0, 1, 32, 0);
}
